// Attention_78348793414287
// MI455X (gfx1250) — matrix operand order repaired and checked by reading, not compiled, not run on hardware
//
#include <hip/hip_runtime.h>

typedef _Float16 half4 __attribute__((ext_vector_type(4)));
typedef _Float16 half8 __attribute__((ext_vector_type(8)));
typedef _Float16 v16h  __attribute__((ext_vector_type(16)));
typedef float    v8f   __attribute__((ext_vector_type(8)));
typedef float    f32x4 __attribute__((ext_vector_type(4)));

#define T_SEQ  4096
#define DMODEL 1024
#define NBATCH 4
#define NHEAD  16
#define NHTOT  (NBATCH*NHEAD)          // 64
#define MROWS  (NBATCH*T_SEQ)          // 16384
#define TCHUNK_KV 8                    // t-chunks for kv partials (deterministic)
#define TCK (T_SEQ/TCHUNK_KV)          // 512

static __device__ __forceinline__ v16h ld_frag_a(const _Float16* base, int row, int g) {
    // 16-bit A 16x32 layout: lane group g=0: K=0..7 & 16..23 ; g=1: K=8..15 & 24..31
    half8 lo = *(const half8*)(base + row*40 + g*8);
    half8 hi = *(const half8*)(base + row*40 + 16 + g*8);
    return __builtin_shufflevector(lo, hi, 0,1,2,3,4,5,6,7,8,9,10,11,12,13,14,15);
}
static __device__ __forceinline__ v16h ld_frag_b40(const _Float16* base, int row, int g) {
    // 16-bit B 32x16: lane column n, lane group g=0: K=0..7 & 16..23 ; g=1: K=8..15 & 24..31 (B staged n-major)
    half8 lo = *(const half8*)(base + row*40 + g*8);
    half8 hi = *(const half8*)(base + row*40 + 16 + g*8);
    return __builtin_shufflevector(lo, hi, 0,1,2,3,4,5,6,7,8,9,10,11,12,13,14,15);
}
static __device__ __forceinline__ v16h ld_frag_b72(const _Float16* base, int row, int k0, int g) {
    half8 lo = *(const half8*)(base + row*72 + k0 + g*8);
    half8 hi = *(const half8*)(base + row*72 + k0 + 16 + g*8);
    return __builtin_shufflevector(lo, hi, 0,1,2,3,4,5,6,7,8,9,10,11,12,13,14,15);
}

// ---------------- Projection GEMM: Out[16384,1024] = X[16384,1024] @ W[1024,1024]
// f16 WMMA, f32 acc, double-buffered LDS (one barrier per k-step), global prefetch.
__global__ __launch_bounds__(128)
void gemm_xw_f16(const float* __restrict__ X, const float* __restrict__ W,
                 float* __restrict__ Out) {
    constexpr unsigned K = 1024, N = 1024;
    constexpr int NSTEP = K / 32;                   // 32
    __shared__ __align__(16) _Float16 As[2][128*40];   // [m][k]
    __shared__ __align__(16) _Float16 Bs[2][128*40];   // [n][k] transposed
    const int tid  = threadIdx.x;
    const int lane = tid & 31, wv = tid >> 5;
    const int lm = lane & 15, g = lane >> 4;
    const unsigned m0 = blockIdx.y * 128, n0 = blockIdx.x * 128;
    const int wM = (wv >> 1) * 64, wN = (wv & 1) * 64;
    const int arow0 = tid >> 3, akc = (tid & 7) * 4;   // A staging coords
    const int bkk0 = tid >> 5, bnc = tid & 31;         // B staging coords

    auto stage = [&](int it, int buf) {
        const unsigned k0 = (unsigned)it * 32u;
        _Float16* a = As[buf];
        _Float16* b = Bs[buf];
#pragma unroll
        for (int i = 0; i < 8; ++i) {               // A tile 128x32 (f32x4 -> half4 store)
            int row = arow0 + i*16;
            f32x4 xv = *(const f32x4*)(X + (m0+row)*K + k0 + akc);
            half4 h = { (_Float16)xv[0], (_Float16)xv[1], (_Float16)xv[2], (_Float16)xv[3] };
            *(half4*)(a + row*40 + akc) = h;
        }
#pragma unroll
        for (int i = 0; i < 8; ++i) {               // W tile 32x128 -> transposed
            int kk = bkk0 + i*4;
            f32x4 wv4 = *(const f32x4*)(W + (k0+kk)*N + n0 + bnc*4);
#pragma unroll
            for (int j = 0; j < 4; ++j) b[(bnc*4+j)*40 + kk] = (_Float16)wv4[j];
        }
    };

    const v8f vz = {0.f,0.f,0.f,0.f,0.f,0.f,0.f,0.f};
    v8f acc[4][4];
#pragma unroll
    for (int i = 0; i < 4; ++i)
#pragma unroll
        for (int j = 0; j < 4; ++j) acc[i][j] = vz;

    stage(0, 0);
    __syncthreads();
    for (int it = 0; it < NSTEP; ++it) {
        const int cur = it & 1;
        if (it + 1 < NSTEP) stage(it + 1, cur ^ 1);
        if (it + 2 < NSTEP) {                        // global_prefetch_b8 for tile k+2
            __builtin_prefetch(X + (m0 + arow0)*K + (unsigned)(it+2)*32u + akc, 0, 1);
            __builtin_prefetch(W + ((unsigned)(it+2)*32u + bkk0)*N + n0 + bnc*4, 0, 1);
        }
        const _Float16* a_s = As[cur];
        const _Float16* b_s = Bs[cur];
        v16h a[4], b[4];
#pragma unroll
        for (int mt = 0; mt < 4; ++mt) a[mt] = ld_frag_a(a_s, wM + mt*16 + lm, g);
#pragma unroll
        for (int nt = 0; nt < 4; ++nt) b[nt] = ld_frag_b40(b_s, wN + nt*16 + lm, g);
#pragma unroll
        for (int mt = 0; mt < 4; ++mt)
#pragma unroll
            for (int nt = 0; nt < 4; ++nt)
                acc[mt][nt] = __builtin_amdgcn_wmma_f32_16x16x32_f16(
                    false, a[mt], false, b[nt], (short)0, acc[mt][nt], false, false);
        __syncthreads();
    }
#pragma unroll
    for (int mt = 0; mt < 4; ++mt)
#pragma unroll
        for (int nt = 0; nt < 4; ++nt)
#pragma unroll
            for (int j = 0; j < 8; ++j)
                Out[(m0 + wM + mt*16 + g*8 + j)*N + n0 + wN + nt*16 + lm]
                    = acc[mt][nt][j];
}

// ---------------- K-softmax column stats (online), chunked over T
__global__ __launch_bounds__(256)
void k_colstats_partial(const float* __restrict__ Kp, const int* __restrict__ mask_attn,
                        float* __restrict__ pmax, float* __restrict__ psum) {
    int col = blockIdx.x * 256 + threadIdx.x;        // 0..4095  (n*1024 + d)
    int n = col >> 10, d = col & 1023;
    int t0 = blockIdx.y * 256;
    float m = -3.0e38f, s = 0.f;
    for (int tt = 0; tt < 256; ++tt) {
        int t = t0 + tt;
        float x = Kp[(unsigned)(n*T_SEQ + t)*DMODEL + d]
                - 10000.f * (1.f - (float)mask_attn[n*T_SEQ + t]);
        if (x > m) { s = s * __expf(m - x) + 1.f; m = x; }
        else       { s += __expf(x - m); }
    }
    pmax[blockIdx.y * 4096 + col] = m;
    psum[blockIdx.y * 4096 + col] = s;
}

__global__ __launch_bounds__(256)
void k_colstats_combine(const float* __restrict__ pmax, const float* __restrict__ psum,
                        float* __restrict__ colM, float* __restrict__ colR) {
    int col = blockIdx.x * 256 + threadIdx.x;
    float m = -3.0e38f;
    for (int c = 0; c < 16; ++c) m = fmaxf(m, pmax[c*4096 + col]);
    float s = 0.f;
    for (int c = 0; c < 16; ++c) s += psum[c*4096 + col] * __expf(pmax[c*4096 + col] - m);
    colM[col] = m;
    colR[col] = 1.f / s;
}

// ---------------- kv partials: kvp[tc][nh][d][e] = sum_t ksm[t][d]*v[t][e] over t-chunk
__global__ __launch_bounds__(128)
void k_kv(const float* __restrict__ Kp, const float* __restrict__ Vp,
          const int* __restrict__ mask_attn,
          const float* __restrict__ colM, const float* __restrict__ colR,
          float* __restrict__ kvp) {
    __shared__ __align__(16) _Float16 Ks[64*40];    // [d][t] (A = ksm^T)
    __shared__ __align__(16) _Float16 Vs[64*40];    // [e][t] (B staged n-major)
    const int tid = threadIdx.x;
    const int nh = blockIdx.x, tc = blockIdx.y;
    const int n = nh >> 4, h = nh & 15;
    const int lane = tid & 31, wv = tid >> 5, lm = lane & 15, g = lane >> 4;
    const int d_s = tid & 63;
    const int col = n*DMODEL + h*64 + d_s;
    const float cM = colM[col], cR = colR[col];

    const v8f vz = {0.f,0.f,0.f,0.f,0.f,0.f,0.f,0.f};
    v8f acc[4] = {vz, vz, vz, vz};

    for (int t0 = tc*TCK; t0 < (tc+1)*TCK; t0 += 32) {
#pragma unroll
        for (int i = 0; i < 16; ++i) {
            int t = (tid >> 6) + i*2;            // 0..31
            unsigned ridx = (unsigned)(n*T_SEQ + t0 + t)*DMODEL + h*64 + d_s;
            float madd = -10000.f * (1.f - (float)mask_attn[n*T_SEQ + t0 + t]);
            Ks[d_s*40 + t] = (_Float16)(__expf(Kp[ridx] + madd - cM) * cR);
            Vs[d_s*40 + t] = (_Float16)Vp[ridx];
        }
        __syncthreads();
        v16h a = ld_frag_a(Ks, wv*16 + lm, g);   // wave owns 16 d-rows
#pragma unroll
        for (int nt = 0; nt < 4; ++nt) {
            v16h b = ld_frag_b40(Vs, nt*16 + lm, g);
            acc[nt] = __builtin_amdgcn_wmma_f32_16x16x32_f16(
                false, a, false, b, (short)0, acc[nt], false, false);
        }
        __syncthreads();
    }
    float* dst = kvp + (unsigned)tc*(NHTOT*4096) + (unsigned)nh*4096;
#pragma unroll
    for (int nt = 0; nt < 4; ++nt)
#pragma unroll
        for (int j = 0; j < 8; ++j)
            dst[(wv*16 + g*8 + j)*64 + nt*16 + lm] = acc[nt][j];
}

// ---------------- out: row-softmax(Qp head slice) @ kv  (in-place over d_out staging)
__global__ __launch_bounds__(128)
void k_out(float* __restrict__ QpOut, const float* __restrict__ kvp) {
    __shared__ __align__(16) float    Qs32[128*65];
    __shared__ __align__(16) _Float16 Qh[128*72];   // softmaxed, f16, [t][d]
    __shared__ __align__(16) _Float16 Kvt[64*72];   // kv transposed, [e][d]
    const int tid = threadIdx.x;
    const int tb = blockIdx.x * 128;
    const int nh = blockIdx.y, n = nh >> 4, h = nh & 15;
    const int lane = tid & 31, wv = tid >> 5, lm = lane & 15, g = lane >> 4;
    const int d_s = tid & 63;

#pragma unroll 4
    for (int i = 0; i < 64; ++i) {                  // stage Qp tile 128x64 (coalesced)
        int r = (tid >> 6) + i*2;
        Qs32[r*65 + d_s] = QpOut[(unsigned)(n*T_SEQ + tb + r)*DMODEL + h*64 + d_s];
    }
#pragma unroll 4
    for (int i = 0; i < 32; ++i) {                  // sum kv partials, stage transposed
        int idx = tid + i*128;                      // d*64+e
        int dd = idx >> 6, e = idx & 63;
        float s = 0.f;
#pragma unroll
        for (int c = 0; c < TCHUNK_KV; ++c)
            s += kvp[(unsigned)c*(NHTOT*4096) + (unsigned)nh*4096 + idx];
        Kvt[e*72 + dd] = (_Float16)s;
    }
    __syncthreads();
    {   // row softmax over dh=64 (mask_q adds a row-uniform constant => exactly invariant)
        int r = tid;
        float m = -3.0e38f;
        for (int d = 0; d < 64; ++d) m = fmaxf(m, Qs32[r*65 + d]);
        float s = 0.f;
        for (int d = 0; d < 64; ++d) s += __expf(Qs32[r*65 + d] - m);
        float rc = 1.f / s;
        for (int d = 0; d < 64; ++d)
            Qh[r*72 + d] = (_Float16)(__expf(Qs32[r*65 + d] - m) * rc);
    }
    __syncthreads();
    const v8f vz = {0.f,0.f,0.f,0.f,0.f,0.f,0.f,0.f};
    v8f acc[2][4];
#pragma unroll
    for (int mt = 0; mt < 2; ++mt)
#pragma unroll
        for (int nt = 0; nt < 4; ++nt) acc[mt][nt] = vz;
#pragma unroll
    for (int ks = 0; ks < 2; ++ks) {                // K = 64 = 2 x 32
        int k0 = ks*32;
        v16h a[2], b[4];
#pragma unroll
        for (int mt = 0; mt < 2; ++mt) {
            int r = wv*32 + mt*16 + lm;
            half8 lo = *(const half8*)(Qh + r*72 + k0 + g*8);
            half8 hi = *(const half8*)(Qh + r*72 + k0 + 16 + g*8);
            a[mt] = __builtin_shufflevector(lo, hi, 0,1,2,3,4,5,6,7,8,9,10,11,12,13,14,15);
        }
#pragma unroll
        for (int nt = 0; nt < 4; ++nt) b[nt] = ld_frag_b72(Kvt, nt*16 + lm, k0, g);
#pragma unroll
        for (int mt = 0; mt < 2; ++mt)
#pragma unroll
            for (int nt = 0; nt < 4; ++nt)
                acc[mt][nt] = __builtin_amdgcn_wmma_f32_16x16x32_f16(
                    false, a[mt], false, b[nt], (short)0, acc[mt][nt], false, false);
    }
#pragma unroll
    for (int mt = 0; mt < 2; ++mt)
#pragma unroll
        for (int nt = 0; nt < 4; ++nt)
#pragma unroll
            for (int j = 0; j < 8; ++j)
                QpOut[(unsigned)(n*T_SEQ + tb + wv*32 + mt*16 + g*8 + j)*DMODEL
                      + h*64 + nt*16 + lm] = acc[mt][nt][j];
}

extern "C" void kernel_launch(void* const* d_in, const int* in_sizes, int n_in,
                              void* d_out, int out_size, void* d_ws, size_t ws_size,
                              hipStream_t stream) {
    (void)in_sizes; (void)n_in; (void)out_size; (void)ws_size;
    const float* q  = (const float*)d_in[0];
    const float* k  = (const float*)d_in[1];
    const float* v  = (const float*)d_in[2];
    const float* Wq = (const float*)d_in[3];
    const float* Wk = (const float*)d_in[4];
    const float* Wv = (const float*)d_in[5];
    // d_in[6] = mask_q: softmax over dh is invariant to row-uniform additive mask -> unused
    const int* mask_attn = (const int*)d_in[7];
    float* out = (float*)d_out;

    char* ws = (char*)d_ws;
    float* Kp   = (float*)(ws);                      // 64 MB
    float* Vp   = (float*)(ws + 67108864);           // 64 MB
    float* kvp  = (float*)(ws + 134217728);          // 8 MB (8 partial chunks)
    float* pmax = (float*)(ws + 142606336);          // 256 KB
    float* psum = (float*)(ws + 142868480);          // 256 KB
    float* colM = (float*)(ws + 143130624);          // 16 KB
    float* colR = (float*)(ws + 143147008);          // 16 KB

    dim3 gg(DMODEL/128, MROWS/128);                  // (8, 128)
    gemm_xw_f16<<<gg, 128, 0, stream>>>(q, Wq, out); // Qp staged in d_out
    gemm_xw_f16<<<gg, 128, 0, stream>>>(k, Wk, Kp);
    gemm_xw_f16<<<gg, 128, 0, stream>>>(v, Wv, Vp);

    k_colstats_partial<<<dim3(16, 16), 256, 0, stream>>>(Kp, mask_attn, pmax, psum);
    k_colstats_combine<<<16, 256, 0, stream>>>(pmax, psum, colM, colR);

    k_kv<<<dim3(NHTOT, TCHUNK_KV), 128, 0, stream>>>(Kp, Vp, mask_attn, colM, colR, kvp);
    k_out<<<dim3(T_SEQ/128, NHTOT), 128, 0, stream>>>(out, kvp);
}
